// BinarizedLinearBlock_120259085186
// MI455X (gfx1250) — matrix operand order repaired and checked by reading, not compiled, not run on hardware
//
#include <hip/hip_runtime.h>

#define B_DIM   16384
#define IN_DIM  2048
#define OUT_DIM 2048
#define BN_EPS  1e-5f

#define TM 128   // batch rows per block
#define TN 128   // output cols per block
#define TK 32    // K per WMMA step

typedef __bf16 bf16x16 __attribute__((ext_vector_type(16)));
typedef __bf16 bf16x8  __attribute__((ext_vector_type(8)));
typedef float  f32x8   __attribute__((ext_vector_type(8)));

__device__ __forceinline__ unsigned short f32_to_bf16_rne(float f) {
    union { float f; unsigned int u; } v; v.f = f;
    unsigned int u = v.u;
    u += 0x7FFFu + ((u >> 16) & 1u);   // round-to-nearest-even
    return (unsigned short)(u >> 16);
}

// ---------------------------------------------------------------------------
// Kernel 1: zero the column sum / sumsq accumulators
// ---------------------------------------------------------------------------
__global__ void zero_ws_kernel(float* __restrict__ p, int n) {
    int i = blockIdx.x * blockDim.x + threadIdx.x;
    if (i < n) p[i] = 0.0f;
}

// ---------------------------------------------------------------------------
// Pre-pass A: x fp32 -> bf16 (memory-bound, one pass; result is L2-resident)
// ---------------------------------------------------------------------------
__global__ __launch_bounds__(256)
void convert_x_kernel(const float* __restrict__ x, unsigned short* __restrict__ xbf) {
    const size_t i    = (size_t)blockIdx.x * blockDim.x + threadIdx.x;
    const size_t base = i * 4;
    const float4 v = *reinterpret_cast<const float4*>(&x[base]);
    ushort4 o;
    o.x = f32_to_bf16_rne(v.x);
    o.y = f32_to_bf16_rne(v.y);
    o.z = f32_to_bf16_rne(v.z);
    o.w = f32_to_bf16_rne(v.w);
    *reinterpret_cast<ushort4*>(&xbf[base]) = o;
}

// ---------------------------------------------------------------------------
// Pre-pass B: sign(w) -> bf16 {+1, -1}
// ---------------------------------------------------------------------------
__global__ __launch_bounds__(256)
void binarize_w_kernel(const float* __restrict__ w, unsigned short* __restrict__ wbf) {
    const size_t i    = (size_t)blockIdx.x * blockDim.x + threadIdx.x;
    const size_t base = i * 4;
    const float4 v = *reinterpret_cast<const float4*>(&w[base]);
    ushort4 o;
    o.x = (v.x >= 0.0f) ? 0x3F80u : 0xBF80u;
    o.y = (v.y >= 0.0f) ? 0x3F80u : 0xBF80u;
    o.z = (v.z >= 0.0f) ? 0x3F80u : 0xBF80u;
    o.w = (v.w >= 0.0f) ? 0x3F80u : 0xBF80u;
    *reinterpret_cast<ushort4*>(&wbf[base]) = o;
}

// ---------------------------------------------------------------------------
// Kernel 2: binarized GEMM via bf16 WMMA + per-column sum/sumsq accumulation
//   grid = (OUT/TN, B/TM), block = 256 threads (8 waves, 4(M) x 2(N) grid)
//   PRE=true : read pre-converted bf16 tiles (no per-tile VALU conversion)
//   PRE=false: convert fp32 -> bf16 / binarize inline (workspace fallback)
// ---------------------------------------------------------------------------
template <bool PRE>
__global__ __launch_bounds__(256)
void gemm_bin_wmma_kernel(const float* __restrict__ x,
                          const float* __restrict__ w,
                          const unsigned short* __restrict__ xbf,
                          const unsigned short* __restrict__ wbf,
                          float* __restrict__ y,
                          float* __restrict__ colsum,
                          float* __restrict__ colsumsq)
{
    // LDS: A tile 128x32 bf16 (row-major, stride TK) = 8KB
    //      B tile 128x32 bf16 (o-major rows, K contiguous) = 8KB
    __shared__ __align__(16) unsigned short As[TM * TK];
    __shared__ __align__(16) unsigned short Bs[TN * TK];

    const int tid   = threadIdx.x;
    const int wave  = tid >> 5;
    const int lane  = tid & 31;
    const int row16 = lane & 15;   // M (A) or N (B/C) within a 16-tile
    const int khalf = lane >> 4;   // which K-half this lane group holds

    const int wm = wave & 3;       // wave M strip: 32 rows each
    const int wn = wave >> 2;      // wave N strip: 64 cols each

    const int blockN = blockIdx.x * TN;
    const int blockM = blockIdx.y * TM;

    f32x8 acc[2][4];
    #pragma unroll
    for (int mt = 0; mt < 2; ++mt)
        #pragma unroll
        for (int nt = 0; nt < 4; ++nt)
            #pragma unroll
            for (int i = 0; i < 8; ++i) acc[mt][nt][i] = 0.0f;

    for (int k0 = 0; k0 < IN_DIM; k0 += TK) {
        __syncthreads();

        if constexpr (PRE) {
            // bf16 tiles: 8KB each = 512 x 16B; 2 slots per thread per tile
            #pragma unroll
            for (int s = 0; s < 2; ++s) {
                const int slot = tid + s * 256;
                const int r  = slot >> 2;        // 0..127
                const int cg = (slot & 3) * 8;   // bf16 column group
                *reinterpret_cast<uint4*>(&As[r * TK + cg]) =
                    *reinterpret_cast<const uint4*>(&xbf[(size_t)(blockM + r) * IN_DIM + k0 + cg]);
                *reinterpret_cast<uint4*>(&Bs[r * TK + cg]) =
                    *reinterpret_cast<const uint4*>(&wbf[(size_t)(blockN + r) * IN_DIM + k0 + cg]);
            }
        } else {
            const int ldr = tid >> 3;        // 0..31
            const int ldc = (tid & 7) * 4;   // float4 granule
            #pragma unroll
            for (int rr = 0; rr < 4; ++rr) {
                const int r = ldr + rr * 32;
                const float4 va = *reinterpret_cast<const float4*>(
                    &x[(size_t)(blockM + r) * IN_DIM + k0 + ldc]);
                unsigned short* da = &As[r * TK + ldc];
                da[0] = f32_to_bf16_rne(va.x);
                da[1] = f32_to_bf16_rne(va.y);
                da[2] = f32_to_bf16_rne(va.z);
                da[3] = f32_to_bf16_rne(va.w);
                const float4 vb = *reinterpret_cast<const float4*>(
                    &w[(size_t)(blockN + r) * IN_DIM + k0 + ldc]);
                unsigned short* db = &Bs[r * TK + ldc];
                db[0] = (vb.x >= 0.0f) ? 0x3F80u : 0xBF80u;
                db[1] = (vb.y >= 0.0f) ? 0x3F80u : 0xBF80u;
                db[2] = (vb.z >= 0.0f) ? 0x3F80u : 0xBF80u;
                db[3] = (vb.w >= 0.0f) ? 0x3F80u : 0xBF80u;
            }
        }
        __syncthreads();

        // ---- load ALL fragments first (single dscnt wait), then 8 WMMAs ----
        bf16x16 afrag[2];
        #pragma unroll
        for (int mt = 0; mt < 2; ++mt) {
            // A fragment 16x32: lane = row (within tile), K = khalf*8+{0..7}, +16
            const unsigned short* ap =
                &As[(wm * 32 + mt * 16 + row16) * TK + khalf * 8];
            const bf16x8 alo = *reinterpret_cast<const bf16x8*>(ap);
            const bf16x8 ahi = *reinterpret_cast<const bf16x8*>(ap + 16);
            #pragma unroll
            for (int i = 0; i < 8; ++i) { afrag[mt][i] = alo[i]; afrag[mt][i + 8] = ahi[i]; }
        }
        bf16x16 bfrag[4];
        #pragma unroll
        for (int nt = 0; nt < 4; ++nt) {
            // B fragment 32x16: lane = column, K = khalf*8+{0..7}, +16
            const unsigned short* bp =
                &Bs[(wn * 64 + nt * 16 + row16) * TK + khalf * 8];
            const bf16x8 blo = *reinterpret_cast<const bf16x8*>(bp);
            const bf16x8 bhi = *reinterpret_cast<const bf16x8*>(bp + 16);
            #pragma unroll
            for (int i = 0; i < 8; ++i) { bfrag[nt][i] = blo[i]; bfrag[nt][i + 8] = bhi[i]; }
        }

        #pragma unroll
        for (int mt = 0; mt < 2; ++mt)
            #pragma unroll
            for (int nt = 0; nt < 4; ++nt)
                acc[mt][nt] = __builtin_amdgcn_wmma_f32_16x16x32_bf16(
                    false, afrag[mt], false, bfrag[nt], (short)0, acc[mt][nt], false, false);
    }

    // ---- epilogue: store y, accumulate column sums / sumsq ----
    #pragma unroll
    for (int mt = 0; mt < 2; ++mt) {
        #pragma unroll
        for (int nt = 0; nt < 4; ++nt) {
            const int col = blockN + wn * 64 + nt * 16 + row16;
            float s = 0.0f, s2 = 0.0f;
            #pragma unroll
            for (int r = 0; r < 8; ++r) {
                const float val = acc[mt][nt][r];
                const int row = blockM + wm * 32 + mt * 16 + khalf * 8 + r;
                y[(size_t)row * OUT_DIM + col] = val;
                s  += val;
                s2 += val * val;
            }
            s  += __shfl_xor(s, 16, 32);
            s2 += __shfl_xor(s2, 16, 32);
            if (khalf == 0) {
                atomicAdd(&colsum[col],   s);
                atomicAdd(&colsumsq[col], s2);
            }
        }
    }
}

// ---------------------------------------------------------------------------
// Kernel 3: fold batch stats into per-column scale/shift
// ---------------------------------------------------------------------------
__global__ void stats_kernel(const float* __restrict__ colsum,
                             const float* __restrict__ colsumsq,
                             const float* __restrict__ gamma,
                             const float* __restrict__ beta,
                             float* __restrict__ scale,
                             float* __restrict__ shift)
{
    int c = blockIdx.x * blockDim.x + threadIdx.x;
    if (c < OUT_DIM) {
        const float invB = 1.0f / (float)B_DIM;
        const float mean = colsum[c] * invB;
        const float var  = colsumsq[c] * invB - mean * mean;
        const float sc   = gamma[c] * rsqrtf(var + BN_EPS);
        scale[c] = sc;
        shift[c] = beta[c] - mean * sc;
    }
}

// ---------------------------------------------------------------------------
// Kernel 4: in-place BN apply + ReLU, float4-vectorized
// ---------------------------------------------------------------------------
__global__ __launch_bounds__(256)
void bnrelu_kernel(float* __restrict__ y,
                   const float* __restrict__ scale,
                   const float* __restrict__ shift)
{
    const size_t i4   = (size_t)blockIdx.x * blockDim.x + threadIdx.x;
    const size_t base = i4 * 4;
    const int col = (int)(base & (OUT_DIM - 1));   // OUT_DIM is a power of two
    float4 v = *reinterpret_cast<float4*>(&y[base]);
    v.x = fmaxf(v.x * scale[col + 0] + shift[col + 0], 0.0f);
    v.y = fmaxf(v.y * scale[col + 1] + shift[col + 1], 0.0f);
    v.z = fmaxf(v.z * scale[col + 2] + shift[col + 2], 0.0f);
    v.w = fmaxf(v.w * scale[col + 3] + shift[col + 3], 0.0f);
    *reinterpret_cast<float4*>(&y[base]) = v;
}

// ---------------------------------------------------------------------------
extern "C" void kernel_launch(void* const* d_in, const int* in_sizes, int n_in,
                              void* d_out, int out_size, void* d_ws, size_t ws_size,
                              hipStream_t stream)
{
    (void)in_sizes; (void)n_in; (void)out_size;

    const float* x     = (const float*)d_in[0];
    const float* w     = (const float*)d_in[1];
    const float* gamma = (const float*)d_in[2];
    const float* beta  = (const float*)d_in[3];
    float* y  = (float*)d_out;
    float* ws = (float*)d_ws;

    float* colsum   = ws;
    float* colsumsq = ws + OUT_DIM;
    float* scale    = ws + 2 * OUT_DIM;
    float* shift    = ws + 3 * OUT_DIM;

    const size_t statsBytes = (size_t)4 * OUT_DIM * sizeof(float);
    const size_t xbfBytes   = (size_t)B_DIM * IN_DIM * sizeof(unsigned short);
    const size_t wbfBytes   = (size_t)OUT_DIM * IN_DIM * sizeof(unsigned short);
    const bool   preconvert = ws_size >= statsBytes + xbfBytes + wbfBytes;

    unsigned short* xbf = (unsigned short*)((char*)d_ws + statsBytes);
    unsigned short* wbf = (unsigned short*)((char*)d_ws + statsBytes + xbfBytes);

    // 1) zero accumulators (deterministic every call)
    zero_ws_kernel<<<(2 * OUT_DIM + 255) / 256, 256, 0, stream>>>(ws, 2 * OUT_DIM);

    dim3 grid(OUT_DIM / TN, B_DIM / TM);
    if (preconvert) {
        // 1b) one-shot conversions; bf16 x (67MB) stays L2-resident for the GEMM
        convert_x_kernel<<<(unsigned)((size_t)B_DIM * IN_DIM / 4 / 256), 256, 0, stream>>>(x, xbf);
        binarize_w_kernel<<<(unsigned)((size_t)OUT_DIM * IN_DIM / 4 / 256), 256, 0, stream>>>(w, wbf);
        gemm_bin_wmma_kernel<true><<<grid, 256, 0, stream>>>(
            x, w, xbf, wbf, y, colsum, colsumsq);
    } else {
        gemm_bin_wmma_kernel<false><<<grid, 256, 0, stream>>>(
            x, w, xbf, wbf, y, colsum, colsumsq);
    }

    // 3) fold stats
    stats_kernel<<<(OUT_DIM + 255) / 256, 256, 0, stream>>>(
        colsum, colsumsq, gamma, beta, scale, shift);

    // 4) BN + ReLU in place
    const size_t nvec = (size_t)B_DIM * OUT_DIM / 4;
    bnrelu_kernel<<<(unsigned)(nvec / 256), 256, 0, stream>>>(y, scale, shift);
}
